// AimTransformer_76227079569798
// MI455X (gfx1250) — compile-verified, matrix operand order checked by reading, not run on hardware
//
#include <hip/hip_runtime.h>
#include <hip/hip_bf16.h>
#include <math.h>

// ---------------------------------------------------------------------------
// AimTransformer for MI455X (gfx1250).  All heavy GEMMs run on
// v_wmma_f32_16x16x32_f16 (wave32).  Each wave computes a 32x64 C macro-tile:
// 2 A fragments + 4 B fragments -> 8 WMMAs per K-step (register-level reuse,
// 8 independent accumulator chains).  Conv stem / LN / attention are VALU.
// ---------------------------------------------------------------------------

typedef _Float16 f16;
typedef __attribute__((ext_vector_type(16))) _Float16 v16h;
typedef __attribute__((ext_vector_type(8)))  float    v8f;

#define BB   8
#define TT   32
#define PP   16
#define DD   384
#define NHH  6
#define HDD  64
#define SS   513      // T*P + 1
#define MSEQ 4104     // B*S
#define MPAD 4128     // 129*32, padded rows for 32-row WMMA macro-tiles

__device__ __forceinline__ float gelu_f(float x) {
    return 0.5f * x * (1.0f + erff(x * 0.70710678118654752f));
}

// CDNA5 16-bit fragment layout: lane = (row|col)&15, lane>>4 selects K half,
// pair p covers k = (p<4 ? 0 : 16) + half*8 + (p&3)*2 (consecutive pair).
__device__ __forceinline__ v16h load_frag(const float* __restrict__ rowptr,
                                          int k0, int half) {
    v16h f;
#pragma unroll
    for (int p = 0; p < 8; ++p) {
        const int kk = k0 + ((p < 4) ? 0 : 16) + half * 8 + ((p & 3) << 1);
        f[2 * p]     = (f16)rowptr[kk];
        f[2 * p + 1] = (f16)rowptr[kk + 1];
    }
    return f;
}

// ---------------------------------------------------------------------------
// WMMA GEMM:  C[M,N] = act(A[M,K] @ W[N,K]^T + bias) (+ R[M,N])
// Row-major.  M%32==0, N%64==0, K%32==0.
// Block = 128 threads = 4 waves, one 32x64 macro-tile per wave.
// ---------------------------------------------------------------------------
template<int ACT, bool HAS_BIAS, bool HAS_RES>
__global__ __launch_bounds__(128)
void gemm_wmma_kernel(const float* __restrict__ A, const float* __restrict__ W,
                      const float* __restrict__ bias, const float* R,
                      float* C, int M, int N, int K)
{
    const int lane   = threadIdx.x & 31;
    const int wave   = threadIdx.x >> 5;
    const int tilesN = N >> 6;                       // 64-wide strips
    const int tile   = blockIdx.x * 4 + wave;
    const int totTiles = (M >> 5) * tilesN;
    if (tile >= totTiles) return;                    // wave-uniform exit
    const int tm = tile / tilesN;
    const int tn = tile % tilesN;
    const int row  = lane & 15;
    const int half = lane >> 4;

    const float* Ar0 = A + (size_t)(tm * 32 + row) * K;
    const float* Ar1 = Ar0 + (size_t)16 * K;
    const float* Wr  = W + (size_t)(tn * 64 + row) * K;

    v8f acc[2][4] = {};
    for (int k0 = 0; k0 < K; k0 += 32) {
        if (k0 + 32 < K) {   // prefetch next K block (global_prefetch_b8)
            __builtin_prefetch(Ar0 + k0 + 32, 0, 3);
            __builtin_prefetch(Ar1 + k0 + 32, 0, 3);
        }
        const v16h a0 = load_frag(Ar0, k0, half);
        const v16h a1 = load_frag(Ar1, k0, half);
#pragma unroll
        for (int j = 0; j < 4; ++j) {
            const v16h bf = load_frag(Wr + (size_t)(j * 16) * K, k0, half);
            acc[0][j] = __builtin_amdgcn_wmma_f32_16x16x32_f16(
                            false, a0, false, bf, (short)0, acc[0][j], false, false);
            acc[1][j] = __builtin_amdgcn_wmma_f32_16x16x32_f16(
                            false, a1, false, bf, (short)0, acc[1][j], false, false);
        }
    }

#pragma unroll
    for (int mi = 0; mi < 2; ++mi) {
#pragma unroll
        for (int j = 0; j < 4; ++j) {
            const int n = tn * 64 + j * 16 + row;
#pragma unroll
            for (int r = 0; r < 8; ++r) {
                const int m = tm * 32 + mi * 16 + r + 8 * half;
                float v = acc[mi][j][r];
                if (HAS_BIAS) v += bias[n];
                if (ACT)      v  = gelu_f(v);
                if (HAS_RES)  v += R[(size_t)m * N + n];
                C[(size_t)m * N + n] = v;
            }
        }
    }
}

// ---------------------------------------------------------------------------
// Conv stem (VALU):  conv + BN + GELU
// ---------------------------------------------------------------------------
__global__ __launch_bounds__(256)
void conv1_kernel(const float* __restrict__ X, const float* __restrict__ w,
                  const float* __restrict__ g, const float* __restrict__ bb,
                  const float* __restrict__ m, const float* __restrict__ v,
                  float* __restrict__ Y)
{
    const int idx = blockIdx.x * 256 + threadIdx.x;
    if (idx >= 256 * 32 * 64 * 64) return;
    const int ox = idx & 63, oy = (idx >> 6) & 63, c = (idx >> 12) & 31, n = idx >> 17;
    const float* img = X + (size_t)n * 128 * 128;
    const float* wk  = w + c * 25;
    const int iy0 = oy * 2 - 2, ix0 = ox * 2 - 2;
    float acc = 0.f;
    for (int ky = 0; ky < 5; ++ky) {
        const int iy = iy0 + ky;
        if ((unsigned)iy < 128u)
            for (int kx = 0; kx < 5; ++kx) {
                const int ix = ix0 + kx;
                if ((unsigned)ix < 128u) acc += img[iy * 128 + ix] * wk[ky * 5 + kx];
            }
    }
    const float inv = g[c] * rsqrtf(v[c] + 1e-5f);
    Y[idx] = gelu_f(acc * inv + (bb[c] - m[c] * inv));
}

__global__ __launch_bounds__(256)
void conv2_kernel(const float* __restrict__ X, const float* __restrict__ w,
                  const float* __restrict__ g, const float* __restrict__ bb,
                  const float* __restrict__ m, const float* __restrict__ v,
                  float* __restrict__ Y)
{
    const int idx = blockIdx.x * 256 + threadIdx.x;
    if (idx >= 256 * 64 * 32 * 32) return;
    const int ox = idx & 31, oy = (idx >> 5) & 31, c = (idx >> 10) & 63, n = idx >> 16;
    const float* xb = X + (size_t)n * 32 * 64 * 64;
    const float* wb = w + c * 32 * 9;
    const int iy0 = oy * 2 - 1, ix0 = ox * 2 - 1;
    float acc = 0.f;
    for (int ci = 0; ci < 32; ++ci) {
        const float* xc = xb + ci * 4096;
        const float* wc = wb + ci * 9;
        for (int ky = 0; ky < 3; ++ky) {
            const int iy = iy0 + ky;
            if ((unsigned)iy < 64u)
                for (int kx = 0; kx < 3; ++kx) {
                    const int ix = ix0 + kx;
                    if ((unsigned)ix < 64u) acc += xc[iy * 64 + ix] * wc[ky * 3 + kx];
                }
        }
    }
    const float inv = g[c] * rsqrtf(v[c] + 1e-5f);
    Y[idx] = gelu_f(acc * inv + (bb[c] - m[c] * inv));
}

__global__ __launch_bounds__(256)
void conv3_kernel(const float* __restrict__ X, const float* __restrict__ w,
                  const float* __restrict__ g, const float* __restrict__ bb,
                  const float* __restrict__ m, const float* __restrict__ v,
                  float* __restrict__ Y)
{
    const int idx = blockIdx.x * 256 + threadIdx.x;
    if (idx >= 256 * 128 * 16 * 16) return;
    const int ox = idx & 15, oy = (idx >> 4) & 15, c = (idx >> 8) & 127, n = idx >> 15;
    const float* xb = X + (size_t)n * 64 * 32 * 32;
    const float* wb = w + c * 64 * 9;
    const int iy0 = oy * 2 - 1, ix0 = ox * 2 - 1;
    float acc = 0.f;
    for (int ci = 0; ci < 64; ++ci) {
        const float* xc = xb + ci * 1024;
        const float* wc = wb + ci * 9;
        for (int ky = 0; ky < 3; ++ky) {
            const int iy = iy0 + ky;
            if ((unsigned)iy < 32u)
                for (int kx = 0; kx < 3; ++kx) {
                    const int ix = ix0 + kx;
                    if ((unsigned)ix < 32u) acc += xc[iy * 32 + ix] * wc[ky * 3 + kx];
                }
        }
    }
    const float inv = g[c] * rsqrtf(v[c] + 1e-5f);
    Y[idx] = gelu_f(acc * inv + (bb[c] - m[c] * inv));
}

// 4x4 average pool -> rows [(n*16+p), c] for the projection GEMM
__global__ __launch_bounds__(256)
void pool_kernel(const float* __restrict__ X, float* __restrict__ Y)
{
    const int idx = blockIdx.x * 256 + threadIdx.x;
    if (idx >= 256 * 16 * 128) return;
    const int c = idx & 127, p = (idx >> 7) & 15, n = idx >> 11;
    const int ph = p >> 2, pw = p & 3;
    const float* src = X + ((size_t)(n * 128 + c) * 16 + ph * 4) * 16 + pw * 4;
    float s = 0.f;
    for (int i = 0; i < 4; ++i)
        for (int j = 0; j < 4; ++j) s += src[i * 16 + j];
    Y[idx] = s * (1.f / 16.f);
}

// (b,t,p,D) -> (b,p,t,D)
__global__ __launch_bounds__(256)
void permute_kernel(const float* __restrict__ X, float* __restrict__ Y)
{
    const int idx = blockIdx.x * 256 + threadIdx.x;
    if (idx >= 4096 * DD) return;
    const int d = idx % DD;
    const int r = idx / DD;          // (b*32+t)*16+p
    const int p = r & 15;
    const int nn = r >> 4;
    const int t = nn & 31;
    const int b = nn >> 5;
    Y[((size_t)((b * 16 + p) * 32 + t)) * DD + d] = X[idx];
}

// depthwise temporal conv (k=3, dilation dil) + bias + GELU, (bp,t,D) layout
__global__ __launch_bounds__(256)
void dwconv_kernel(const float* __restrict__ X, const float* __restrict__ w,
                   const float* __restrict__ bias, float* __restrict__ H, int dil)
{
    const int idx = blockIdx.x * 256 + threadIdx.x;
    if (idx >= 4096 * DD) return;
    const int d = idx % DD;
    const int r = idx / DD;
    const int t  = r & 31;
    const int bp = r >> 5;
    float acc = bias[d];
    for (int k = 0; k < 3; ++k) {
        const int tt = t + (k - 1) * dil;
        if ((unsigned)tt < 32u)
            acc += w[d * 3 + k] * X[((size_t)(bp * 32 + tt)) * DD + d];
    }
    H[idx] = gelu_f(acc);
}

// ---------------------------------------------------------------------------
// LayerNorm over D=384, one block (128 thr) per row
// ---------------------------------------------------------------------------
__global__ __launch_bounds__(128)
void ln_kernel(const float* __restrict__ X, const float* __restrict__ g,
               const float* __restrict__ b, float* __restrict__ Y)
{
    __shared__ float xr[DD];
    __shared__ float red[128];
    const int tid = threadIdx.x;
    const float* x = X + (size_t)blockIdx.x * DD;
    float s = 0.f;
    for (int i = tid; i < DD; i += 128) { const float t = x[i]; xr[i] = t; s += t; }
    red[tid] = s; __syncthreads();
    for (int st = 64; st; st >>= 1) { if (tid < st) red[tid] += red[tid + st]; __syncthreads(); }
    const float mu = red[0] * (1.f / DD);
    __syncthreads();
    float s2 = 0.f;
    for (int i = tid; i < DD; i += 128) { const float d = xr[i] - mu; s2 += d * d; }
    red[tid] = s2; __syncthreads();
    for (int st = 64; st; st >>= 1) { if (tid < st) red[tid] += red[tid + st]; __syncthreads(); }
    const float inv = rsqrtf(red[0] * (1.f / DD) + 1e-5f);
    float* y = Y + (size_t)blockIdx.x * DD;
    for (int i = tid; i < DD; i += 128) y[i] = (xr[i] - mu) * inv * g[i] + b[i];
}

// tnorm(out2 + tc) + positions + CLS -> x_seq ; one block per (b,s)
__global__ __launch_bounds__(128)
void assemble_kernel(const float* __restrict__ out2, const float* __restrict__ tcb,
                     const float* __restrict__ tg, const float* __restrict__ tb,
                     const float* __restrict__ fpos, const float* __restrict__ ppos,
                     const float* __restrict__ clst, const float* __restrict__ clsp,
                     float* __restrict__ xseq)
{
    __shared__ float xr[DD];
    __shared__ float red[128];
    const int tid = threadIdx.x;
    const int bs = blockIdx.x;
    const int b = bs / SS, s = bs % SS;
    float* y = xseq + (size_t)bs * DD;
    if (s == 0) {
        for (int i = tid; i < DD; i += 128) y[i] = clst[i] + clsp[i];
        return;
    }
    const int t = (s - 1) >> 4, p = (s - 1) & 15;
    const size_t r = ((size_t)(b * 16 + p) * 32 + t) * DD;
    float sm = 0.f;
    for (int i = tid; i < DD; i += 128) { const float v = out2[r + i] + tcb[r + i]; xr[i] = v; sm += v; }
    red[tid] = sm; __syncthreads();
    for (int st = 64; st; st >>= 1) { if (tid < st) red[tid] += red[tid + st]; __syncthreads(); }
    const float mu = red[0] * (1.f / DD);
    __syncthreads();
    float s2 = 0.f;
    for (int i = tid; i < DD; i += 128) { const float d = xr[i] - mu; s2 += d * d; }
    red[tid] = s2; __syncthreads();
    for (int st = 64; st; st >>= 1) { if (tid < st) red[tid] += red[tid + st]; __syncthreads(); }
    const float inv = rsqrtf(red[0] * (1.f / DD) + 1e-5f);
    for (int i = tid; i < DD; i += 128)
        y[i] = (xr[i] - mu) * inv * tg[i] + tb[i] + fpos[t * DD + i] + ppos[p * DD + i];
}

// ---------------------------------------------------------------------------
// Fused masked softmax attention: one block per (b, head, q)
// qkv rows: [q(384) | k(384) | v(384)]
// mask: q>=1 attends to k>=1 with frame(k) >= frame(q); k=0 masked; q=0 full.
// ---------------------------------------------------------------------------
__global__ __launch_bounds__(128)
void attn_kernel(const float* __restrict__ qkv, float* __restrict__ o)
{
    __shared__ float sc[SS];
    __shared__ float red[128];
    __shared__ float qv[HDD];
    int id = blockIdx.x;
    const int q = id % SS; id /= SS;
    const int h = id % NHH;
    const int b = id / NHH;
    const int tid = threadIdx.x;
    const int hc = h * HDD;
    const size_t base = (size_t)b * SS;

    for (int i = tid; i < HDD; i += 128) qv[i] = qkv[(base + q) * 1152 + hc + i];
    __syncthreads();

    const int qf = (q == 0) ? -1 : ((q - 1) >> 4);
    float lmax = -3.4e38f;
    for (int k = tid; k < SS; k += 128) {
        bool allowed = (q == 0) ? true : (k != 0 && ((k - 1) >> 4) >= qf);
        float sv = -3.4e38f;
        if (allowed) {
            const float* kr = qkv + (base + k) * 1152 + 384 + hc;
            float d = 0.f;
            for (int j = 0; j < HDD; ++j) d += qv[j] * kr[j];
            sv = d * 0.125f;  // 1/sqrt(64)
        }
        sc[k] = sv;
        lmax = fmaxf(lmax, sv);
    }
    red[tid] = lmax; __syncthreads();
    for (int st = 64; st; st >>= 1) { if (tid < st) red[tid] = fmaxf(red[tid], red[tid + st]); __syncthreads(); }
    const float mx = red[0];
    __syncthreads();
    float lsum = 0.f;
    for (int k = tid; k < SS; k += 128) { const float e = expf(sc[k] - mx); sc[k] = e; lsum += e; }
    red[tid] = lsum; __syncthreads();
    for (int st = 64; st; st >>= 1) { if (tid < st) red[tid] += red[tid + st]; __syncthreads(); }
    const float inv = 1.f / red[0];
    __syncthreads();

    const int d = tid & 63, part = tid >> 6;
    float acc = 0.f;
    for (int k = part; k < SS; k += 2)
        acc += sc[k] * qkv[(base + k) * 1152 + 768 + hc + d];
    red[tid] = acc; __syncthreads();
    if (part == 0)
        o[(base + q) * DD + hc + d] = (red[d] + red[d + 64]) * inv;
}

// ---------------------------------------------------------------------------
// Head: LN + gelu(W1) then sigmoid(W2)
// ---------------------------------------------------------------------------
__global__ __launch_bounds__(128)
void head1_kernel(const float* __restrict__ xseq, const float* __restrict__ lg,
                  const float* __restrict__ lb, const float* __restrict__ w1,
                  const float* __restrict__ b1, float* __restrict__ hout)
{
    __shared__ float xr[DD];
    __shared__ float red[128];
    const int tid = threadIdx.x;
    const int b = blockIdx.x;
    const float* x = xseq + (size_t)b * SS * DD;  // row s=0 (CLS)
    float s = 0.f;
    for (int i = tid; i < DD; i += 128) { const float t = x[i]; xr[i] = t; s += t; }
    red[tid] = s; __syncthreads();
    for (int st = 64; st; st >>= 1) { if (tid < st) red[tid] += red[tid + st]; __syncthreads(); }
    const float mu = red[0] * (1.f / DD);
    __syncthreads();
    float s2 = 0.f;
    for (int i = tid; i < DD; i += 128) { const float d = xr[i] - mu; s2 += d * d; }
    red[tid] = s2; __syncthreads();
    for (int st = 64; st; st >>= 1) { if (tid < st) red[tid] += red[tid + st]; __syncthreads(); }
    const float inv = rsqrtf(red[0] * (1.f / DD) + 1e-5f);
    __syncthreads();
    for (int i = tid; i < DD; i += 128) xr[i] = (xr[i] - mu) * inv * lg[i] + lb[i];
    __syncthreads();
    for (int o = tid; o < DD; o += 128) {
        const float* wr = w1 + (size_t)o * DD;
        float a = b1[o];
        for (int j = 0; j < DD; ++j) a += xr[j] * wr[j];
        hout[b * DD + o] = gelu_f(a);
    }
}

__global__ void head2_kernel(const float* __restrict__ hbuf, const float* __restrict__ w2,
                             const float* __restrict__ b2, float* __restrict__ out)
{
    const int idx = blockIdx.x * blockDim.x + threadIdx.x;
    if (idx >= 16) return;
    const int b = idx >> 1, j = idx & 1;
    const float* hh = hbuf + b * DD;
    const float* wr = w2 + j * DD;
    float a = b2[j];
    for (int k = 0; k < DD; ++k) a += hh[k] * wr[k];
    out[idx] = 1.f / (1.f + expf(-a));
}

// ---------------------------------------------------------------------------
extern "C" void kernel_launch(void* const* d_in, const int* in_sizes, int n_in,
                              void* d_out, int out_size, void* d_ws, size_t ws_size,
                              hipStream_t stream)
{
    (void)in_sizes; (void)n_in; (void)out_size; (void)ws_size;
    const float* frames   = (const float*)d_in[0];
    const float* conv1_w  = (const float*)d_in[1];
    const float* bn1_g    = (const float*)d_in[2];
    const float* bn1_b    = (const float*)d_in[3];
    const float* bn1_m    = (const float*)d_in[4];
    const float* bn1_v    = (const float*)d_in[5];
    const float* conv2_w  = (const float*)d_in[6];
    const float* bn2_g    = (const float*)d_in[7];
    const float* bn2_b    = (const float*)d_in[8];
    const float* bn2_m    = (const float*)d_in[9];
    const float* bn2_v    = (const float*)d_in[10];
    const float* conv3_w  = (const float*)d_in[11];
    const float* bn3_g    = (const float*)d_in[12];
    const float* bn3_b    = (const float*)d_in[13];
    const float* bn3_m    = (const float*)d_in[14];
    const float* bn3_v    = (const float*)d_in[15];
    const float* proj_w   = (const float*)d_in[16];
    const float* dw_w     = (const float*)d_in[17];
    const float* dw_b     = (const float*)d_in[18];
    const float* pw_w     = (const float*)d_in[19];
    const float* pw_b     = (const float*)d_in[20];
    const float* tnorm_g  = (const float*)d_in[21];
    const float* tnorm_b  = (const float*)d_in[22];
    const float* frame_pos= (const float*)d_in[23];
    const float* patch_pos= (const float*)d_in[24];
    const float* cls_tok  = (const float*)d_in[25];
    const float* cls_pos  = (const float*)d_in[26];
    const float* qkv_w    = (const float*)d_in[27];
    const float* qkv_b    = (const float*)d_in[28];
    const float* out_w    = (const float*)d_in[29];
    const float* out_b    = (const float*)d_in[30];
    const float* ln1_g    = (const float*)d_in[31];
    const float* ln1_b    = (const float*)d_in[32];
    const float* ln2_g    = (const float*)d_in[33];
    const float* ln2_b    = (const float*)d_in[34];
    const float* ffn_w1   = (const float*)d_in[35];
    const float* ffn_b1   = (const float*)d_in[36];
    const float* ffn_w2   = (const float*)d_in[37];
    const float* ffn_b2   = (const float*)d_in[38];
    const float* hln_g    = (const float*)d_in[39];
    const float* hln_b    = (const float*)d_in[40];
    const float* head_w1  = (const float*)d_in[41];
    const float* head_b1  = (const float*)d_in[42];
    const float* head_w2  = (const float*)d_in[43];
    const float* head_b2  = (const float*)d_in[44];

    float* ws = (float*)d_ws;
    // Region 0: conv1 out (33.55M f), later reused for sequence buffers
    float* conv1o = ws;
    // Region 1: conv2 out (16.78M f), later reused for token/temporal buffers
    float* conv2o = ws + 33554432;
    // Region 2: conv3 out (8.39M f)
    float* conv3o = ws + 50331648;
    // R1 reuse (after conv3 done)
    float* Apool  = conv2o;                  //  524288
    float* tok    = conv2o + 524288;         // 1572864  (b,t,p,D)
    float* tcb    = conv2o + 2097152;        // 1572864  (b,p,t,D)
    float* htc    = conv2o + 3670016;        // 1572864
    float* out1   = conv2o + 5242880;        // 1572864
    float* out2   = conv2o + 6815744;        // 1572864
    float* headh  = conv2o + 8388608;        //    3072
    // R0 reuse (after conv2 done); rows padded to MPAD=4128
    float* xseq   = conv1o;                  // 4128*384  = 1585152
    float* hseq   = conv1o + 1585152;        // 4128*384
    float* attno  = conv1o + 3170304;        // 4128*384
    float* qkvb   = conv1o + 4755456;        // 4128*1152 = 4755456
    float* ffnh   = conv1o + 9510912;        // 4128*1536 = 6340608 -> end 15851520

    auto gemm = [&](const float* A, const float* W, const float* bias, const float* R,
                    float* C, int M, int N, int K, int act) {
        const int grid = ((M / 32) * (N / 64) + 3) / 4;
        if (act)              gemm_wmma_kernel<1, true,  false><<<grid, 128, 0, stream>>>(A, W, bias, R, C, M, N, K);
        else if (bias && R)   gemm_wmma_kernel<0, true,  true ><<<grid, 128, 0, stream>>>(A, W, bias, R, C, M, N, K);
        else if (bias)        gemm_wmma_kernel<0, true,  false><<<grid, 128, 0, stream>>>(A, W, bias, R, C, M, N, K);
        else                  gemm_wmma_kernel<0, false, false><<<grid, 128, 0, stream>>>(A, W, bias, R, C, M, N, K);
    };

    // ---- conv stem ----
    conv1_kernel<<<131072, 256, 0, stream>>>(frames, conv1_w, bn1_g, bn1_b, bn1_m, bn1_v, conv1o);
    conv2_kernel<<< 65536, 256, 0, stream>>>(conv1o, conv2_w, bn2_g, bn2_b, bn2_m, bn2_v, conv2o);
    conv3_kernel<<< 32768, 256, 0, stream>>>(conv2o, conv3_w, bn3_g, bn3_b, bn3_m, bn3_v, conv3o);

    // ---- pool + projection (WMMA GEMM 4096x384x128) ----
    pool_kernel<<<2048, 256, 0, stream>>>(conv3o, Apool);
    gemm(Apool, proj_w, nullptr, nullptr, tok, 4096, DD, 128, 0);

    // ---- temporal conv module ----
    permute_kernel<<<6144, 256, 0, stream>>>(tok, tcb);
    dwconv_kernel<<<6144, 256, 0, stream>>>(tcb, dw_w,            dw_b,        htc, 1);
    gemm(htc, pw_w,            pw_b,        tcb,  out1, 4096, DD, DD, 0);
    dwconv_kernel<<<6144, 256, 0, stream>>>(out1, dw_w + DD * 3,  dw_b + DD,   htc, 2);
    gemm(htc, pw_w + DD * DD,  pw_b + DD,   out1, out2, 4096, DD, DD, 0);

    // ---- tnorm + positions + CLS -> sequence ----
    assemble_kernel<<<MSEQ, 128, 0, stream>>>(out2, tcb, tnorm_g, tnorm_b,
                                              frame_pos, patch_pos, cls_tok, cls_pos, xseq);

    // ---- transformer layers ----
    for (int i = 0; i < 6; ++i) {
        ln_kernel<<<MSEQ, 128, 0, stream>>>(xseq, ln1_g + i * DD, ln1_b + i * DD, hseq);
        gemm(hseq, qkv_w + (size_t)i * 1152 * DD, qkv_b + i * 1152, nullptr, qkvb, MPAD, 1152, DD, 0);
        attn_kernel<<<BB * NHH * SS, 128, 0, stream>>>(qkvb, attno);
        gemm(attno, out_w + (size_t)i * DD * DD, out_b + i * DD, xseq, xseq, MPAD, DD, DD, 0);
        ln_kernel<<<MSEQ, 128, 0, stream>>>(xseq, ln2_g + i * DD, ln2_b + i * DD, hseq);
        gemm(hseq, ffn_w1 + (size_t)i * 1536 * DD, ffn_b1 + i * 1536, nullptr, ffnh, MPAD, 1536, DD, 1);
        gemm(ffnh, ffn_w2 + (size_t)i * DD * 1536, ffn_b2 + i * DD, xseq, xseq, MPAD, DD, 1536, 0);
    }

    // ---- head ----
    head1_kernel<<<BB, 128, 0, stream>>>(xseq, hln_g, hln_b, head_w1, head_b1, headh);
    head2_kernel<<<1, 32, 0, stream>>>(headh, head_w2, head_b2, (float*)d_out);
}
